// GraphLSTM_16784732192960
// MI455X (gfx1250) — matrix operand order repaired and checked by reading, not compiled, not run on hardware
//
#include <hip/hip_runtime.h>

// ---------------- types ----------------
typedef __attribute__((ext_vector_type(16))) __bf16    v16bf;
typedef __attribute__((ext_vector_type(8)))  float     v8f;
typedef __attribute__((ext_vector_type(4)))  unsigned  u32x4;
typedef __attribute__((ext_vector_type(4)))  float     f32x4;

#define T_STEPS 8
#define NROWS_N 50000
#define NROWS_E 100000
#define NROWS_G 512
#define D 128
#define TILES_N (NROWS_N/16)   // 3125
#define TILES_E (NROWS_E/16)   // 6250
#define TILES_G (NROWS_G/16)   // 32
#define TILES_TOTAL (TILES_N + TILES_E + TILES_G) // 9407

// f32 -> bf16 round-to-nearest-even (returned in low 16 bits)
__device__ __forceinline__ unsigned f2bf_u(float f) {
  unsigned u = __builtin_bit_cast(unsigned, f);
  return (u + 0x7FFFu + ((u >> 16) & 1u)) >> 16;
}

__device__ __forceinline__ float fast_tanhf(float x) {
#if __has_builtin(__builtin_amdgcn_tanhf)
  return __builtin_amdgcn_tanhf(x);          // V_TANH_F32 (CDNA5 trans op)
#else
  float e = __expf(2.0f * x);
  return (e - 1.0f) / (e + 1.0f);
#endif
}
__device__ __forceinline__ float fast_sigmoid(float x) {
  return 0.5f * fast_tanhf(0.5f * x) + 0.5f;
}

__device__ __forceinline__ v8f wmma_bf16(v16bf a, v16bf b, v8f c) {
  // D = A(16x32 bf16) * B(32x16 bf16) + C(16x16 f32)
  return __builtin_amdgcn_wmma_f32_16x16x32_bf16(
      /*neg_a=*/false, a, /*neg_b=*/false, b,
      /*c_mod=*/(short)0, c, /*reuse_a=*/false, /*reuse_b=*/false);
}

struct Q2 { u32x4 a, b; };
__device__ __forceinline__ v16bf make_frag(u32x4 a, u32x4 b) {
  Q2 q; q.a = a; q.b = b;
  return __builtin_bit_cast(v16bf, q);
}

// Pre-packed B fragment: 256 u32 words, lane reads words [lane*8, lane*8+8)
__device__ __forceinline__ v16bf load_bfrag(const unsigned* p, int lane) {
  const u32x4* q = (const u32x4*)(p + lane * 8);
  return make_frag(q[0], q[1]);
}

// A fragment (16x32 bf16) from an LDS tile with row stride 68 u32 (136 halfwords).
// ISA layout: lanes 0-15: V0-3 <- K 0..7, V4-7 <- K 16..23 (row M = lane);
//             lanes 16-31: V0-3 <- K 8..15, V4-7 <- K 24..31 (row M = lane-16).
__device__ __forceinline__ v16bf load_afrag(const unsigned* buf, int lane, int kt) {
  int r   = lane & 15;
  int off = (lane < 16) ? 0 : 4;               // u32 offset (8 halfwords)
  const unsigned* p = buf + r * 68 + kt * 16 + off;
  u32x4 a = *(const u32x4*)(p);
  u32x4 b = *(const u32x4*)(p + 8);
  return make_frag(a, b);
}

// ---------------- weight pre-pack kernel ----------------
// pw layout: comp c at c*32768 u32; frag f (W1: 0-63, W2: 64-95, W3: 96-127)
// at f*256; word w = lane*8 + v. V_v of lane l holds bf16 pair
// (K0, K0+1) with K0 = kt*32 + 2v + (v<4 ? 0 : 8) + (l<16 ? 0 : 8), N = nt*16 + (l&15).
__global__ void pack_weights_kernel(
    const float* W1n, const float* W2n, const float* W3n,
    const float* W1e, const float* W2e, const float* W3e,
    const float* W1g, const float* W2g, const float* W3g,
    unsigned* pw)
{
  int gid = blockIdx.x * blockDim.x + threadIdx.x;
  if (gid >= 3 * 32768) return;
  int c    = gid >> 15;
  int rem  = gid & 32767;
  int frag = rem >> 8;
  int w    = rem & 255;
  int l    = w >> 3;
  int v    = w & 7;
  const float* W;
  int kt, nt;
  if (frag < 64) {
    W = (c == 0) ? W1n : ((c == 1) ? W1e : W1g);
    kt = frag >> 3; nt = frag & 7;                 // K total 256
  } else if (frag < 96) {
    W = (c == 0) ? W2n : ((c == 1) ? W2e : W2g);
    int f = frag - 64; kt = f >> 3; nt = f & 7;    // K total 128
  } else {
    W = (c == 0) ? W3n : ((c == 1) ? W3e : W3g);
    int f = frag - 96; kt = f >> 3; nt = f & 7;
  }
  int N  = nt * 16 + (l & 15);
  int K0 = kt * 32 + 2 * v + ((v < 4) ? 0 : 8) + ((l < 16) ? 0 : 8);
  unsigned lo = f2bf_u(W[(size_t)K0 * D + N]);
  unsigned hi = f2bf_u(W[(size_t)(K0 + 1) * D + N]);
  pw[gid] = lo | (hi << 16);
}

// ---------------- main kernel: 1 wave per block, one 16-row tile ----------------
__global__ void __launch_bounds__(32, 1) graph_lstm_kernel(
    const float* __restrict__ xs, const float* __restrict__ es, const float* __restrict__ gs,
    const float* __restrict__ b1n, const float* __restrict__ b2n, const float* __restrict__ b3n,
    const float* __restrict__ b1e, const float* __restrict__ b2e, const float* __restrict__ b3e,
    const float* __restrict__ b1g, const float* __restrict__ b2g, const float* __restrict__ b3g,
    const unsigned* __restrict__ pw, float* __restrict__ out)
{
  __shared__ __align__(16) float    ldsF[16 * 132]; // f32 x tile, padded stride
  __shared__ __align__(16) unsigned ldsX[16 * 68];  // bf16 x tile (u32 pairs)
  __shared__ __align__(16) unsigned ldsO[16 * 68];  // bf16 out tile

  const int lane = threadIdx.x;
  const int tile = blockIdx.x;
  if (tile >= TILES_TOTAL) return;

  int c, lt;
  const float *X, *B1, *B2, *B3;
  long long Mrows;
  size_t stBase;
  if (tile < TILES_N) {
    c = 0; lt = tile; X = xs; B1 = b1n; B2 = b2n; B3 = b3n;
    Mrows = NROWS_N; stBase = 0;
  } else if (tile < TILES_N + TILES_E) {
    c = 1; lt = tile - TILES_N; X = es; B1 = b1e; B2 = b2e; B3 = b3e;
    Mrows = NROWS_E; stBase = (size_t)NROWS_N * D;
  } else {
    c = 2; lt = tile - (TILES_N + TILES_E); X = gs; B1 = b1g; B2 = b2g; B3 = b3g;
    Mrows = NROWS_G; stBase = (size_t)(NROWS_N + NROWS_E) * D;
  }
  size_t outBase = (size_t)(NROWS_N + NROWS_E + NROWS_G) * D + stBase;
  const int row0 = lt * 16;

  // biases (per-lane column values, 8 n-tiles each)
  float b1v[8], b2v[8], b3v[8];
#pragma unroll
  for (int nt = 0; nt < 8; ++nt) {
    b1v[nt] = B1[nt * 16 + (lane & 15)];
    b2v[nt] = B2[nt * 16 + (lane & 15)];
    b3v[nt] = B3[nt * 16 + (lane & 15)];
  }

  // ---- load tile t=0, also initializing out := x_0 (bf16) ----
  {
    int r = lane >> 1, c0 = (lane & 1) << 6;   // lane covers half a row
    const float* srcrow = X + ((size_t)row0 + r) * D + c0;
#pragma unroll
    for (int j = 0; j < 16; ++j) {
      f32x4 f = __builtin_nontemporal_load((const f32x4*)(srcrow + j * 4));
      int col = c0 + j * 4;
      *(f32x4*)&ldsF[r * 132 + col] = f;
      unsigned p0 = f2bf_u(f.x) | (f2bf_u(f.y) << 16);
      unsigned p1 = f2bf_u(f.z) | (f2bf_u(f.w) << 16);
      ldsX[r * 68 + (col >> 1)]     = p0;
      ldsX[r * 68 + (col >> 1) + 1] = p1;
      ldsO[r * 68 + (col >> 1)]     = p0;
      ldsO[r * 68 + (col >> 1) + 1] = p1;
    }
  }
  __syncthreads();

  // ---- init state := x_0 (f32, C-layout registers) ----
  float st[64];
  const int h = (lane >> 4) << 3;  // 0 or 8 (row offset of this half-wave in C layout)
#pragma unroll
  for (int nt = 0; nt < 8; ++nt)
#pragma unroll
    for (int r2 = 0; r2 < 8; ++r2)
      st[nt * 8 + r2] = ldsF[(r2 + h) * 132 + nt * 16 + (lane & 15)];

  float* stp  = out + stBase  + (size_t)row0 * D;
  float* outp = out + outBase + (size_t)row0 * D;

#pragma clang loop unroll(disable)
  for (int t = 0; t < T_STEPS; ++t) {
    // Opaque zero offset: defeats LICM so fragment loads are re-issued each
    // timestep (they stay L2/WGP$-resident) instead of being hoisted across
    // the whole recurrence and spilled. Keeping the *pointer* untouched
    // preserves address-space inference -> global_load (not flat_load).
    unsigned wzero = 0;
    asm volatile("" : "+v"(wzero));
    const unsigned* w1 = pw + (size_t)c * 32768 + wzero;
    const unsigned* w2 = w1 + 64 * 256;
    const unsigned* w3 = w1 + 96 * 256;

    // A-fragments of previous out (cat() low half, K 0..127)
    v16bf oA[4];
#pragma unroll
    for (int kt = 0; kt < 4; ++kt) oA[kt] = load_afrag(ldsO, lane, kt);
    __syncthreads();

    if (t > 0) {  // load x_t tile (t==0 already resident)
      int r = lane >> 1, c0 = (lane & 1) << 6;
      const float* srcrow = X + ((size_t)t * Mrows + row0 + r) * D + c0;
#pragma unroll
      for (int j = 0; j < 16; ++j) {
        f32x4 f = __builtin_nontemporal_load((const f32x4*)(srcrow + j * 4));
        int col = c0 + j * 4;
        *(f32x4*)&ldsF[r * 132 + col] = f;
        ldsX[r * 68 + (col >> 1)]     = f2bf_u(f.x) | (f2bf_u(f.y) << 16);
        ldsX[r * 68 + (col >> 1) + 1] = f2bf_u(f.z) | (f2bf_u(f.w) << 16);
      }
      __syncthreads();
    }

    v16bf xA[4];
#pragma unroll
    for (int kt = 0; kt < 4; ++kt) xA[kt] = load_afrag(ldsX, lane, kt);

#pragma unroll
    for (int nt = 0; nt < 8; ++nt) {
      v8f g1 = {0,0,0,0,0,0,0,0};
      v8f g2 = {0,0,0,0,0,0,0,0};
      v8f g3 = {0,0,0,0,0,0,0,0};
#pragma unroll
      for (int kt = 0; kt < 4; ++kt) {
        g2 = wmma_bf16(xA[kt], load_bfrag(w2 + (kt * 8 + nt) * 256, lane), g2);
        g3 = wmma_bf16(xA[kt], load_bfrag(w3 + (kt * 8 + nt) * 256, lane), g3);
        g1 = wmma_bf16(oA[kt], load_bfrag(w1 + (kt * 8 + nt) * 256, lane), g1);       // out half
        g1 = wmma_bf16(xA[kt], load_bfrag(w1 + ((kt + 4) * 8 + nt) * 256, lane), g1); // x half
      }
      int ncol = nt * 16 + (lane & 15);
#pragma unroll
      for (int r2 = 0; r2 < 8; ++r2) {
        float xv = ldsF[(r2 + h) * 132 + ncol];
        float fg = fast_sigmoid(g1[r2] + b1v[nt]);
        float ig = fast_sigmoid(g2[r2] + b2v[nt]);
        float og = fast_sigmoid(g3[r2] + b3v[nt]);
        float s  = st[nt * 8 + r2] * fg + ig * fast_tanhf(xv);
        st[nt * 8 + r2] = s;
        float ov = fast_tanhf(s) * og;
        ((unsigned short*)ldsO)[(r2 + h) * 136 + ncol] = (unsigned short)f2bf_u(ov);
        if (t == T_STEPS - 1) {
          __builtin_nontemporal_store(s,  &stp [(size_t)(r2 + h) * D + ncol]);
          __builtin_nontemporal_store(ov, &outp[(size_t)(r2 + h) * D + ncol]);
        }
      }
    }
    __syncthreads();
  }
}

// ---------------- launcher ----------------
extern "C" void kernel_launch(void* const* d_in, const int* in_sizes, int n_in,
                              void* d_out, int out_size, void* d_ws, size_t ws_size,
                              hipStream_t stream) {
  const float* xs = (const float*)d_in[0];
  const float* es = (const float*)d_in[1];
  const float* gs = (const float*)d_in[2];
  const float* W1n = (const float*)d_in[3];  const float* b1n = (const float*)d_in[4];
  const float* W2n = (const float*)d_in[5];  const float* b2n = (const float*)d_in[6];
  const float* W3n = (const float*)d_in[7];  const float* b3n = (const float*)d_in[8];
  const float* W1e = (const float*)d_in[9];  const float* b1e = (const float*)d_in[10];
  const float* W2e = (const float*)d_in[11]; const float* b2e = (const float*)d_in[12];
  const float* W3e = (const float*)d_in[13]; const float* b3e = (const float*)d_in[14];
  const float* W1g = (const float*)d_in[15]; const float* b1g = (const float*)d_in[16];
  const float* W2g = (const float*)d_in[17]; const float* b2g = (const float*)d_in[18];
  const float* W3g = (const float*)d_in[19]; const float* b3g = (const float*)d_in[20];

  unsigned* pw = (unsigned*)d_ws;  // 3 * 32768 u32 = 384 KB packed bf16 weights

  pack_weights_kernel<<<(3 * 32768 + 255) / 256, 256, 0, stream>>>(
      W1n, W2n, W3n, W1e, W2e, W3e, W1g, W2g, W3g, pw);

  graph_lstm_kernel<<<TILES_TOTAL, 32, 0, stream>>>(
      xs, es, gs,
      b1n, b2n, b3n, b1e, b2e, b3e, b1g, b2g, b3g,
      pw, (float*)d_out);
}
